// LinearTransformerEncoderLayer_28475633172714
// MI455X (gfx1250) — matrix operand order repaired and checked by reading, not compiled, not run on hardware
//
#include <hip/hip_runtime.h>

// ---------------------------------------------------------------------------
// LinearTransformerEncoderLayer on gfx1250 (MI455X):
//   Double-buffered LDS-tiled bf16 WMMA GEMM (v_wmma_f32_16x16x32_bf16),
//   async global->LDS staging (ASYNCcnt), BK=64 (2 compute sub-steps per
//   staged tile), all fragments prefetched before the WMMA chain.
// ---------------------------------------------------------------------------

#define D_MODEL 1024
#define D_FF    4096
#define SEQ_N   2048
#define BATCH   8
#define M_TOT   (BATCH * SEQ_N)   // 16384 rows
#define LN_EPS  1e-5f
#define FM_EPS  1e-6f

#define BM 128
#define BN 128
#define BK 64
#define LDSS 72                   // padded LDS row stride (bf16 elems, 144 B)
#define TILE_ELEMS (BM * LDSS)    // per A/B tile buffer (18432 B)

#define USE_ASYNC 1

typedef __attribute__((ext_vector_type(16))) __bf16 v16bf;
typedef __attribute__((ext_vector_type(8)))  int    v8i;
typedef __attribute__((ext_vector_type(8)))  float  v8f;

union FragU { v8i i; v16bf b16; };

__device__ __forceinline__ unsigned short f2bf(float f) {
    unsigned int u = __float_as_uint(f);
    unsigned int r = u + 0x7FFFu + ((u >> 16) & 1u);   // RNE
    return (unsigned short)(r >> 16);
}
__device__ __forceinline__ float bf2f(unsigned short h) {
    return __uint_as_float(((unsigned int)h) << 16);
}

#if USE_ASYNC
// CDNA5 async path: global -> LDS, tracked by ASYNCcnt.
__device__ __forceinline__ void async_cp_b128(unsigned lds_off, const void* gp) {
    asm volatile("global_load_async_to_lds_b128 %0, %1, off"
                 :: "v"(lds_off), "v"((unsigned long long)(uintptr_t)gp)
                 : "memory");
}
__device__ __forceinline__ void async_wait0() {
    asm volatile("s_wait_asynccnt 0x0" ::: "memory");
}
#endif

// ---------------------------------------------------------------------------
// fp32 -> bf16 convert (grid-stride)
// ---------------------------------------------------------------------------
__global__ void cvt_f32_bf16(const float* __restrict__ in,
                             unsigned short* __restrict__ out, long n) {
    long i = (long)blockIdx.x * blockDim.x + threadIdx.x;
    long stride = (long)gridDim.x * blockDim.x;
    for (; i < n; i += stride) out[i] = f2bf(in[i]);
}

// ---------------------------------------------------------------------------
// Tiled transpose + convert: out[c][r] (bf16, ld=R) = in[r][c] (f32, ld=C)
// ---------------------------------------------------------------------------
__global__ __launch_bounds__(256)
void transpose_cvt(const float* __restrict__ in, unsigned short* __restrict__ out,
                   int R, int C) {
    __shared__ float tile[32][33];
    const int tx = threadIdx.x & 31, ty = threadIdx.x >> 5;
    const int bx = blockIdx.x * 32, by = blockIdx.y * 32;
#pragma unroll
    for (int j = 0; j < 32; j += 8)
        tile[ty + j][tx] = in[(long)(by + ty + j) * C + bx + tx];
    __syncthreads();
#pragma unroll
    for (int j = 0; j < 32; j += 8)
        out[(long)(bx + ty + j) * R + by + tx] = f2bf(tile[tx][ty + j]);
}

// ---------------------------------------------------------------------------
// Tiled bf16 GEMM (TN form): C[m,n] = sum_k A[m,k] * BT[n,k]  (+bias/relu/eps)
//   256 threads = 8 waves; 128x128 tile; BK=64; wave = 32x64 (16 wmma/tile);
//   double-buffered LDS, async global->LDS staging, fragments prefetched.
//   CTM==1: scatter transposed bf16 output CT[b][col][tok] (token rows).
// ---------------------------------------------------------------------------
template<int RELU, bool HASBIAS, bool O16, bool O32, int CTM>
__global__ __launch_bounds__(256)
void gemm_tn(const unsigned short* __restrict__ A, int ldA, long batchA,
             const unsigned short* __restrict__ BT, int ldBT, long batchB,
             const float* __restrict__ bias,
             unsigned short* __restrict__ C16, float* __restrict__ C32,
             unsigned short* __restrict__ CT16,
             long batchC, int ldC, int Kdim, float eps_add)
{
    __shared__ unsigned short smem[2 * 2 * TILE_ELEMS];  // [buf][A|B][...]

    const int tid  = threadIdx.x;
    const int wave = tid >> 5;
    const int lane = tid & 31;
    const int half = lane >> 4;
    const int l15  = lane & 15;

    const int wm = wave & 3;            // 4 row groups * 32 rows
    const int wn = wave >> 2;           // 2 col groups * 64 cols
    const int m0b = blockIdx.y * BM;
    const int n0b = blockIdx.x * BN;
    const long z  = blockIdx.z;

    const unsigned short* Ab  = A  + z * batchA;
    const unsigned short* BTb = BT + z * batchB;

    v8f acc[2][4] = {};

    // staging: per array 128 rows x 64 elems; 4 b128 chunks per thread
    const int srow = tid >> 3;          // 0..31 (+32 steps)
    const int scol = (tid & 7) * 8;     // k element offset

    auto stage = [&](int kk, int buf) {
        const int aoff = buf * 2 * TILE_ELEMS;
        const int boff = aoff + TILE_ELEMS;
#pragma unroll
        for (int r = 0; r < 4; ++r) {
            const int row = srow + r * 32;
            const unsigned short* ga = Ab  + (long)(m0b + row) * ldA  + kk + scol;
            const unsigned short* gb = BTb + (long)(n0b + row) * ldBT + kk + scol;
#if USE_ASYNC
            async_cp_b128((unsigned)(uintptr_t)&smem[aoff + row * LDSS + scol], ga);
            async_cp_b128((unsigned)(uintptr_t)&smem[boff + row * LDSS + scol], gb);
#else
            *(uint4*)&smem[aoff + row * LDSS + scol] = *(const uint4*)ga;
            *(uint4*)&smem[boff + row * LDSS + scol] = *(const uint4*)gb;
#endif
        }
    };

    stage(0, 0);
    const int nk = Kdim >> 6;
    for (int kt = 0; kt < nk; ++kt) {
        const int cur = kt & 1;
#if USE_ASYNC
        async_wait0();                  // buf[cur] staging complete (this wave)
#endif
        __syncthreads();                // all waves' staging visible
        if (kt + 1 < nk) stage((kt + 1) << 6, cur ^ 1);

        const int aoff = cur * 2 * TILE_ELEMS;
        const int boff = aoff + TILE_ELEMS;
#pragma unroll
        for (int ks = 0; ks < BK; ks += 32) {
            // ---- prefetch all fragments for this 32-K sub-step ----
            FragU a[2], b[4];
#pragma unroll
            for (int mi = 0; mi < 2; ++mi) {
                const unsigned short* ap =
                    &smem[aoff + (wm * 32 + mi * 16 + l15) * LDSS + ks + half * 8];
                const uint4 alo = *(const uint4*)(ap);
                const uint4 ahi = *(const uint4*)(ap + 16);
                a[mi].i[0] = (int)alo.x; a[mi].i[1] = (int)alo.y;
                a[mi].i[2] = (int)alo.z; a[mi].i[3] = (int)alo.w;
                a[mi].i[4] = (int)ahi.x; a[mi].i[5] = (int)ahi.y;
                a[mi].i[6] = (int)ahi.z; a[mi].i[7] = (int)ahi.w;
            }
#pragma unroll
            for (int t = 0; t < 4; ++t) {
                const unsigned short* bp =
                    &smem[boff + (wn * 64 + t * 16 + l15) * LDSS + ks + half * 8];
                const uint4 b0 = *(const uint4*)(bp);
                const uint4 b1 = *(const uint4*)(bp + 16);
                b[t].i[0] = (int)b0.x; b[t].i[1] = (int)b0.y;
                b[t].i[2] = (int)b0.z; b[t].i[3] = (int)b0.w;
                b[t].i[4] = (int)b1.x; b[t].i[5] = (int)b1.y;
                b[t].i[6] = (int)b1.z; b[t].i[7] = (int)b1.w;
            }
            // ---- 8 WMMA, B fragment reused across M-subtiles ----
#pragma unroll
            for (int t = 0; t < 4; ++t)
#pragma unroll
                for (int mi = 0; mi < 2; ++mi)
                    acc[mi][t] = __builtin_amdgcn_wmma_f32_16x16x32_bf16(
                        false, a[mi].b16, false, b[t].b16, (short)0,
                        acc[mi][t], false, false);
        }
    }

    // ---- epilogue (templated; dead paths eliminated) ----
    const long cbase = z * batchC;
#pragma unroll
    for (int mi = 0; mi < 2; ++mi) {
#pragma unroll
        for (int t = 0; t < 4; ++t) {
            const int col = n0b + wn * 64 + t * 16 + l15;
            const float badd = HASBIAS ? bias[col] : 0.f;
#pragma unroll
            for (int r = 0; r < 8; ++r) {
                float v = acc[mi][t][r] + badd;
                if (RELU >= 1) v = v > 0.f ? v : 0.f;
                if (RELU == 2) v += eps_add;
                const int row = m0b + wm * 32 + mi * 16 + half * 8 + r;
                if (O32) C32[cbase + (long)row * ldC + col] = v;
                if (O16) C16[cbase + (long)row * ldC + col] = f2bf(v);
                if (CTM == 1) {   // token rows: CT[b][col][tok]
                    const long it = (long)(row >> 11) * ((long)D_MODEL * SEQ_N)
                                  + (long)col * SEQ_N + (row & 2047);
                    CT16[it] = f2bf(v);
                }
            }
        }
    }
}

// ---------------------------------------------------------------------------
// Ksum[b][d] = sum_n KT[b,d,n]  (one wave per contiguous K^T row)
// ---------------------------------------------------------------------------
__global__ __launch_bounds__(256)
void ksum_kernel(const unsigned short* __restrict__ KT, float* __restrict__ Ksum) {
    const int wave = threadIdx.x >> 5, lane = threadIdx.x & 31;
    const int d = blockIdx.x * 8 + wave;
    const int b = blockIdx.y;
    const unsigned short* kp = KT + (long)b * D_MODEL * SEQ_N + (long)d * SEQ_N;
    float s = 0.f;
    for (int n = lane; n < SEQ_N; n += 32) s += bf2f(kp[n]);
#pragma unroll
    for (int o = 16; o > 0; o >>= 1) s += __shfl_xor(s, o, 32);
    if (lane == 0) Ksum[b * D_MODEL + d] = s;
}

// ---------------------------------------------------------------------------
// wave32 block reduction (blockDim.x == 256)
// ---------------------------------------------------------------------------
__device__ __forceinline__ float blockReduceSum(float v, float* sm) {
#pragma unroll
    for (int o = 16; o > 0; o >>= 1) v += __shfl_xor(v, o, 32);
    const int wid = threadIdx.x >> 5, lid = threadIdx.x & 31;
    if (lid == 0) sm[wid] = v;
    __syncthreads();
    float t = (threadIdx.x < 8) ? sm[threadIdx.x] : 0.f;
    if (wid == 0) {
#pragma unroll
        for (int o = 4; o > 0; o >>= 1) t += __shfl_xor(t, o, 32);
        if (lid == 0) sm[0] = t;
    }
    __syncthreads();
    const float r = sm[0];
    __syncthreads();
    return r;
}

// ---------------------------------------------------------------------------
// Per-row: Z = 1/(Q.Ksum + eps);  y = src + attnraw*Z;  x = LN(y)*g1 + be1
// ---------------------------------------------------------------------------
__global__ __launch_bounds__(256)
void attn_ln_kernel(const float* __restrict__ src,
                    const unsigned short* __restrict__ Qb,
                    const float* __restrict__ attnraw,
                    const float* __restrict__ Ksum,
                    const float* __restrict__ g1, const float* __restrict__ be1,
                    float* __restrict__ x, unsigned short* __restrict__ xb) {
    __shared__ float sm[8];
    const long row = blockIdx.x;
    const int  b   = (int)(row >> 11);
    const long base = row * D_MODEL;
    const float* ks = Ksum + b * D_MODEL;

    float dotv = 0.f;
#pragma unroll
    for (int j = 0; j < 4; ++j) {
        const int i = threadIdx.x + j * 256;
        dotv += bf2f(Qb[base + i]) * ks[i];
    }
    dotv = blockReduceSum(dotv, sm);
    const float zval = 1.f / (dotv + FM_EPS);

    float y[4]; float s = 0.f;
#pragma unroll
    for (int j = 0; j < 4; ++j) {
        const int i = threadIdx.x + j * 256;
        y[j] = src[base + i] + attnraw[base + i] * zval;
        s += y[j];
    }
    const float mu = blockReduceSum(s, sm) * (1.f / D_MODEL);
    float vs = 0.f;
#pragma unroll
    for (int j = 0; j < 4; ++j) { const float d = y[j] - mu; vs += d * d; }
    const float var = blockReduceSum(vs, sm) * (1.f / D_MODEL);
    const float inv = rsqrtf(var + LN_EPS);
#pragma unroll
    for (int j = 0; j < 4; ++j) {
        const int i = threadIdx.x + j * 256;
        const float o = (y[j] - mu) * inv * g1[i] + be1[i];
        x[base + i]  = o;
        xb[base + i] = f2bf(o);
    }
}

// ---------------------------------------------------------------------------
// Per-row: out = LN(x + ffraw)*g2 + be2
// ---------------------------------------------------------------------------
__global__ __launch_bounds__(256)
void final_ln_kernel(const float* __restrict__ x,
                     const float* __restrict__ ffraw,
                     const float* __restrict__ g2, const float* __restrict__ be2,
                     float* __restrict__ out) {
    __shared__ float sm[8];
    const long base = (long)blockIdx.x * D_MODEL;
    float y[4]; float s = 0.f;
#pragma unroll
    for (int j = 0; j < 4; ++j) {
        const int i = threadIdx.x + j * 256;
        y[j] = x[base + i] + ffraw[base + i];
        s += y[j];
    }
    const float mu = blockReduceSum(s, sm) * (1.f / D_MODEL);
    float vs = 0.f;
#pragma unroll
    for (int j = 0; j < 4; ++j) { const float d = y[j] - mu; vs += d * d; }
    const float var = blockReduceSum(vs, sm) * (1.f / D_MODEL);
    const float inv = rsqrtf(var + LN_EPS);
#pragma unroll
    for (int j = 0; j < 4; ++j) {
        const int i = threadIdx.x + j * 256;
        out[base + i] = (y[j] - mu) * inv * g2[i] + be2[i];
    }
}

// ---------------------------------------------------------------------------
// Host-side orchestration
// ---------------------------------------------------------------------------
static inline size_t alignup(size_t v) { return (v + 255) & ~(size_t)255; }

extern "C" void kernel_launch(void* const* d_in, const int* in_sizes, int n_in,
                              void* d_out, int out_size, void* d_ws, size_t ws_size,
                              hipStream_t stream) {
    (void)in_sizes; (void)n_in; (void)out_size; (void)ws_size;

    const float* src = (const float*)d_in[0];
    const float* Wq  = (const float*)d_in[1];
    const float* bq  = (const float*)d_in[2];
    const float* Wk  = (const float*)d_in[3];
    const float* bk  = (const float*)d_in[4];
    const float* Wv  = (const float*)d_in[5];
    const float* bv  = (const float*)d_in[6];
    const float* W1  = (const float*)d_in[7];
    const float* b1  = (const float*)d_in[8];
    const float* W2  = (const float*)d_in[9];
    const float* b2  = (const float*)d_in[10];
    const float* g1  = (const float*)d_in[11];
    const float* be1 = (const float*)d_in[12];
    const float* g2  = (const float*)d_in[13];
    const float* be2 = (const float*)d_in[14];
    float* out = (float*)d_out;

    // ---- workspace layout ----
    char* ws = (char*)d_ws;
    size_t off = 0;
    const size_t o_srcb = off; off = alignup(off + (size_t)M_TOT * D_MODEL * 2);
    const size_t o_WqT  = off; off = alignup(off + (size_t)D_MODEL * D_MODEL * 2);
    const size_t o_WkT  = off; off = alignup(off + (size_t)D_MODEL * D_MODEL * 2);
    const size_t o_WvT  = off; off = alignup(off + (size_t)D_MODEL * D_MODEL * 2);
    const size_t o_W1T  = off; off = alignup(off + (size_t)D_MODEL * D_FF * 2);
    const size_t o_W2T  = off; off = alignup(off + (size_t)D_FF * D_MODEL * 2);
    const size_t o_Qb   = off; off = alignup(off + (size_t)M_TOT * D_MODEL * 2);
    const size_t o_KT   = off; off = alignup(off + (size_t)M_TOT * D_MODEL * 2);
    const size_t o_VT   = off; off = alignup(off + (size_t)M_TOT * D_MODEL * 2);
    const size_t o_KVT  = off; off = alignup(off + (size_t)BATCH * D_MODEL * D_MODEL * 2);
    const size_t o_Ksum = off; off = alignup(off + (size_t)BATCH * D_MODEL * 4);
    const size_t o_x    = off; off = alignup(off + (size_t)M_TOT * D_MODEL * 4);
    const size_t o_xb   = off; off = alignup(off + (size_t)M_TOT * D_MODEL * 2);
    const size_t o_hb   = off; off = alignup(off + (size_t)M_TOT * D_FF * 2);
    const size_t o_attn = o_hb;   // attnraw f32 overlays hb (dead before FF1)
    const size_t o_ff   = o_Qb;   // ffraw f32 overlays Qb+KT (dead after attn_ln)

    unsigned short* srcb = (unsigned short*)(ws + o_srcb);
    unsigned short* WqT  = (unsigned short*)(ws + o_WqT);
    unsigned short* WkT  = (unsigned short*)(ws + o_WkT);
    unsigned short* WvT  = (unsigned short*)(ws + o_WvT);
    unsigned short* W1T  = (unsigned short*)(ws + o_W1T);
    unsigned short* W2T  = (unsigned short*)(ws + o_W2T);
    unsigned short* Qb   = (unsigned short*)(ws + o_Qb);
    unsigned short* KT   = (unsigned short*)(ws + o_KT);
    unsigned short* VT   = (unsigned short*)(ws + o_VT);
    unsigned short* KVT  = (unsigned short*)(ws + o_KVT);
    float* Ksum = (float*)(ws + o_Ksum);
    float* x    = (float*)(ws + o_x);
    unsigned short* xb = (unsigned short*)(ws + o_xb);
    unsigned short* hb = (unsigned short*)(ws + o_hb);
    float* attnraw = (float*)(ws + o_attn);
    float* ffraw   = (float*)(ws + o_ff);

    // ---- 1: convert (weights transposed to [n][k] for TN GEMM) ----
    cvt_f32_bf16<<<4096, 256, 0, stream>>>(src, srcb, (long)M_TOT * D_MODEL);
    transpose_cvt<<<dim3(D_MODEL/32, D_MODEL/32), 256, 0, stream>>>(Wq, WqT, D_MODEL, D_MODEL);
    transpose_cvt<<<dim3(D_MODEL/32, D_MODEL/32), 256, 0, stream>>>(Wk, WkT, D_MODEL, D_MODEL);
    transpose_cvt<<<dim3(D_MODEL/32, D_MODEL/32), 256, 0, stream>>>(Wv, WvT, D_MODEL, D_MODEL);
    transpose_cvt<<<dim3(D_FF/32,    D_MODEL/32), 256, 0, stream>>>(W1, W1T, D_MODEL, D_FF);
    transpose_cvt<<<dim3(D_MODEL/32, D_FF/32),    256, 0, stream>>>(W2, W2T, D_FF, D_MODEL);

    // ---- 2: Q/K/V projections (M=16384, N=1024, K=1024) ----
    {
        dim3 grid(D_MODEL / BN, M_TOT / BM, 1);
        gemm_tn<2, true, true, false, 0><<<grid, 256, 0, stream>>>(   // Q
            srcb, D_MODEL, 0, WqT, D_MODEL, 0, bq,
            Qb, nullptr, nullptr, 0, D_MODEL, D_MODEL, FM_EPS);
        gemm_tn<2, true, false, false, 1><<<grid, 256, 0, stream>>>(  // K^T
            srcb, D_MODEL, 0, WkT, D_MODEL, 0, bk,
            nullptr, nullptr, KT, 0, D_MODEL, D_MODEL, FM_EPS);
        gemm_tn<0, true, false, false, 1><<<grid, 256, 0, stream>>>(  // V^T
            srcb, D_MODEL, 0, WvT, D_MODEL, 0, bv,
            nullptr, nullptr, VT, 0, D_MODEL, D_MODEL, 0.f);
    }

    // ---- 3: Ksum[b,d] from K^T ----
    ksum_kernel<<<dim3(D_MODEL / 8, BATCH, 1), 256, 0, stream>>>(KT, Ksum);

    // ---- 4: KV^T[b] = V^T[b] @ K[b]  (A=VT, BT=KT; M=N=1024, K=2048) ----
    {
        dim3 grid(D_MODEL / BN, D_MODEL / BM, BATCH);
        gemm_tn<0, false, true, false, 0><<<grid, 256, 0, stream>>>(
            VT, SEQ_N, (long)D_MODEL * SEQ_N,
            KT, SEQ_N, (long)D_MODEL * SEQ_N,
            nullptr, KVT, nullptr, nullptr,
            (long)D_MODEL * D_MODEL, D_MODEL, SEQ_N, 0.f);
    }

    // ---- 5: attnraw[b] = Q[b] @ KV[b]  (A=Qb, BT=KV^T; f32 out) ----
    {
        dim3 grid(D_MODEL / BN, SEQ_N / BM, BATCH);
        gemm_tn<0, false, false, true, 0><<<grid, 256, 0, stream>>>(
            Qb, D_MODEL, (long)SEQ_N * D_MODEL,
            KVT, D_MODEL, (long)D_MODEL * D_MODEL,
            nullptr, nullptr, attnraw, nullptr,
            (long)SEQ_N * D_MODEL, D_MODEL, D_MODEL, 0.f);
    }

    // ---- 6: Z-scale + residual + LayerNorm1 -> x, xb ----
    attn_ln_kernel<<<M_TOT, 256, 0, stream>>>(src, Qb, attnraw, Ksum, g1, be1, x, xb);

    // ---- 7: FFN hidden h = relu(x@W1 + b1)  (BT=W1T) ----
    {
        dim3 grid(D_FF / BN, M_TOT / BM, 1);
        gemm_tn<1, true, true, false, 0><<<grid, 256, 0, stream>>>(
            xb, D_MODEL, 0, W1T, D_MODEL, 0, b1,
            hb, nullptr, nullptr, 0, D_FF, D_MODEL, 0.f);
    }

    // ---- 8: ffraw = h@W2 + b2  (BT=W2T; f32 out) ----
    {
        dim3 grid(D_MODEL / BN, M_TOT / BM, 1);
        gemm_tn<0, true, false, true, 0><<<grid, 256, 0, stream>>>(
            hb, D_FF, 0, W2T, D_FF, 0, b2,
            nullptr, ffraw, nullptr, 0, D_MODEL, D_FF, 0.f);
    }

    // ---- 9: residual + LayerNorm2 -> out ----
    final_ln_kernel<<<M_TOT, 256, 0, stream>>>(x, ffraw, g2, be2, out);
}
